// SelfAttention_80272938762345
// MI455X (gfx1250) — compile-verified, matrix operand order checked by reading, not run on hardware
//
#include <hip/hip_runtime.h>

typedef __attribute__((ext_vector_type(16))) _Float16 v16h;
typedef __attribute__((ext_vector_type(8)))  _Float16 v8h;
typedef __attribute__((ext_vector_type(8)))  float    v8f;
typedef __attribute__((ext_vector_type(4)))  unsigned u32x4;
typedef __attribute__((ext_vector_type(8)))  int      i32x8;
typedef __attribute__((ext_vector_type(4)))  int      i32x4;

#define BATCH 16
#define HEADS 12
#define SEQ   1024
#define HDIM  64
#define BH    (BATCH*HEADS)

__device__ __forceinline__ v8f wmma_f16(v16h a, v16h b, v8f c) {
  return __builtin_amdgcn_wmma_f32_16x16x32_f16(false, a, false, b, (short)0, c,
                                                false, false);
}

__device__ __forceinline__ v16h cat8(v8h lo, v8h hi_) {
  return __builtin_shufflevector(lo, hi_, 0, 1, 2, 3, 4, 5, 6, 7,
                                 8, 9, 10, 11, 12, 13, 14, 15);
}

// ---------------------------------------------------------------------------
// TDM: 2D tile load Global->LDS.  D# built per ISA 08_async_tensor §8.
// data_size=1 (2-byte elements).  Issued wave-uniform (EXEC ignored by TDM).
// ---------------------------------------------------------------------------
__device__ __forceinline__ void tdm_load_2d(unsigned lds_off, const void* gptr,
                                            unsigned tile_d0, unsigned tile_d1,
                                            unsigned stride_d0) {
  unsigned long long ga = (unsigned long long)gptr;
  u32x4 g0;
  g0[0] = 1u;                                        // count=1, user descriptor
  g0[1] = lds_off;                                   // lds_addr (bytes)
  g0[2] = (unsigned)(ga & 0xFFFFFFFFu);              // global_addr lo
  g0[3] = (unsigned)((ga >> 32) & 0x01FFFFFFu) | (2u << 30);  // addr hi | type=2
  i32x8 g1;
  g1[0] = (int)(1u << 16);                           // wg_mask=0, data_size=1 (2B)
  g1[1] = (int)((tile_d0 & 0xFFFFu) << 16);          // tensor_dim0[15:0]
  g1[2] = (int)(((tile_d0 >> 16) & 0xFFFFu) |        // tensor_dim0[31:16]
                ((tile_d1 & 0xFFFFu) << 16));        // tensor_dim1[15:0]
  g1[3] = (int)(((tile_d1 >> 16) & 0xFFFFu) |        // tensor_dim1[31:16]
                ((tile_d0 & 0xFFFFu) << 16));        // tile_dim0
  g1[4] = (int)(tile_d1 & 0xFFFFu);                  // tile_dim1 (tile_dim2=0)
  g1[5] = (int)stride_d0;                            // tensor_dim0_stride lo32
  g1[6] = 0;
  g1[7] = 0;
  i32x4 z4 = {0, 0, 0, 0};
  i32x8 z8 = {0, 0, 0, 0, 0, 0, 0, 0};
  __builtin_amdgcn_tensor_load_to_lds(g0, g1, z4, z4, z8, 0);
}

// ---------------------------------------------------------------------------
// Kernel 1: per-head QKV projection.  y = x @ W^T + b   (1024x64 @ 64x64 / bh)
// Q pre-scaled by 1/sqrt(64).  Q,K stored row-major f16; V stored TRANSPOSED
// (VhT[bh][d][s]) so kernel 2's P@V B-fragments are contiguous per lane.
// Weights pre-packed in LDS in WMMA B-fragment layout (contiguous per lane).
// ---------------------------------------------------------------------------
__global__ __launch_bounds__(256) void qkv_project_kernel(
    const float* __restrict__ x,
    const float* __restrict__ Wq, const float* __restrict__ bq,
    const float* __restrict__ Wk, const float* __restrict__ bk,
    const float* __restrict__ Wv, const float* __restrict__ bv,
    _Float16* __restrict__ Qh, _Float16* __restrict__ Kh, _Float16* __restrict__ VhT)
{
  // Wp[w][frag = kb*4+nb][lane][16 halves]
  __shared__ __align__(32) _Float16 Wp[3 * 8 * 32 * 16];
  __shared__ float bias[3][HDIM];

  const int tid  = threadIdx.x;
  const int bh   = blockIdx.x >> 3;
  const int tile = blockIdx.x & 7;

  if (tid < HDIM) {
    bias[0][tid] = bq[tid];
    bias[1][tid] = bk[tid];
    bias[2][tid] = bv[tid];
  }
  {
    const float* Ws[3] = {Wq, Wk, Wv};
#pragma unroll
    for (int g3 = 0; g3 < 3; ++g3) {            // 768 fragment-lane groups total
      const int w  = g3;
      const int f  = (tid >> 5) & 7;
      const int L  = tid & 31;
      const int kb = f >> 2, nb = f & 3;
      const int e  = nb * 16 + (L & 15);
      const int db = kb * 32 + ((L >> 4) << 3);
      const float* src = Ws[w] + e * HDIM + db;
      _Float16* dst = Wp + ((w * 8 + f) * 32 + L) * 16;
#pragma unroll
      for (int j = 0; j < 8; ++j) {
        dst[j]     = (_Float16)src[j];          // d = db + j
        dst[8 + j] = (_Float16)src[16 + j];     // d = db + 16 + j
      }
    }
  }
  __syncthreads();

  const int lane = tid & 31;
  const int wave = tid >> 5;
  const int lrow = lane & 15;
  const int hi   = lane >> 4;
  const int koff = hi * 8;
  const int rowBase = tile * 128 + wave * 16;
  const size_t base = (size_t)bh * SEQ * HDIM;

  // A-fragments (16x32) from x, f32 -> f16 on the fly
  const float* xrow = x + base + (size_t)(rowBase + lrow) * HDIM;
  v16h a0, a1;
#pragma unroll
  for (int j = 0; j < 8; ++j) {
    a0[j]     = (_Float16)xrow[koff + j];
    a0[8 + j] = (_Float16)xrow[16 + koff + j];
    a1[j]     = (_Float16)xrow[32 + koff + j];
    a1[8 + j] = (_Float16)xrow[48 + koff + j];
  }

#pragma unroll
  for (int w = 0; w < 3; ++w) {
    const float scale = (w == 0) ? 0.125f : 1.0f;
#pragma unroll
    for (int nb = 0; nb < 4; ++nb) {
      const int col = nb * 16 + lrow;
      v16h b0 = *(const v16h*)(Wp + ((w * 8 + 0 + nb) * 32 + lane) * 16);
      v16h b1 = *(const v16h*)(Wp + ((w * 8 + 4 + nb) * 32 + lane) * 16);
      v8f c = {};
      c = wmma_f16(a0, b0, c);
      c = wmma_f16(a1, b1, c);
      const float bb = bias[w][col];
      if (w == 2) {
        // V: transposed store VhT[d=col][s], contiguous 8 rows per lane -> b128
        union { uint4 u; _Float16 h[8]; } pk;
#pragma unroll
        for (int r = 0; r < 8; ++r) pk.h[r] = (_Float16)(c[r] + bb);
        *(uint4*)(VhT + base + (size_t)col * SEQ + rowBase + 8 * hi) = pk.u;
      } else {
        _Float16* dst = (w == 0 ? Qh : Kh) + base;
#pragma unroll
        for (int r = 0; r < 8; ++r) {
          int m = rowBase + r + 8 * hi;
          dst[(size_t)m * HDIM + col] = (_Float16)((c[r] + bb) * scale);
        }
      }
    }
  }
}

// ---------------------------------------------------------------------------
// Kernel 2: flash attention (online softmax) + fused output projection.
// K/V tiles double-buffered through the TDM: tile k+1 streams into LDS while
// tile k is consumed (TENSORcnt completes in-order -> wait(2) = tile k done).
// ---------------------------------------------------------------------------
__global__ __launch_bounds__(256) void attn_out_kernel(
    const _Float16* __restrict__ Qh, const _Float16* __restrict__ Kh,
    const _Float16* __restrict__ VhT,
    const float* __restrict__ Wo, const float* __restrict__ bo,
    float* __restrict__ out)
{
  __shared__ __align__(16) _Float16 Ktile[2][128 * HDIM];   // 32 KB (buf0 reused)
  __shared__ __align__(16) _Float16 VtileT[2][HDIM * 128];  // 32 KB (dim-major)
  __shared__ __align__(16) _Float16 Pscr[8][16 * 32];       //  8 KB
  __shared__ __align__(32) _Float16 WoP[8 * 32 * 16];       //  8 KB packed B-frags
  __shared__ float bob[HDIM];

  const int tid   = threadIdx.x;
  const int bh    = blockIdx.x >> 3;
  const int qtile = blockIdx.x & 7;

  // stage Wo as packed B-fragments: one (frag,lane) group per thread
  {
    const int f  = tid >> 5;
    const int L  = tid & 31;
    const int kb = f >> 2, nb = f & 3;
    const int e  = nb * 16 + (L & 15);
    const int db = kb * 32 + ((L >> 4) << 3);
    const float* src = Wo + e * HDIM + db;
    _Float16* dst = WoP + (f * 32 + L) * 16;
#pragma unroll
    for (int j = 0; j < 8; ++j) {
      dst[j]     = (_Float16)src[j];
      dst[8 + j] = (_Float16)src[16 + j];
    }
  }
  if (tid < HDIM) bob[tid] = bo[tid];

  const int lane = tid & 31;
  const int wave = tid >> 5;
  const int lrow = lane & 15;
  const int hi   = lane >> 4;
  const size_t base = (size_t)bh * SEQ * HDIM;
  const int qrow = qtile * 128 + wave * 16;

  // Q A-fragments (already scaled by 1/8): contiguous per-lane b128 loads
  v16h aq0, aq1;
  {
    const v8h* qv = (const v8h*)(Qh + base + (size_t)(qrow + lrow) * HDIM);
    aq0 = cat8(qv[hi], qv[2 + hi]);
    aq1 = cat8(qv[4 + hi], qv[6 + hi]);
  }

  v8f oacc[4];
#pragma unroll
  for (int cb = 0; cb < 4; ++cb) oacc[cb] = (v8f){};
  float mrow[8], lsum[8];
#pragma unroll
  for (int r = 0; r < 8; ++r) { mrow[r] = -1e30f; lsum[r] = 0.f; }

  const unsigned ldsK[2] = {(unsigned)(size_t)(void*)&Ktile[0][0],
                            (unsigned)(size_t)(void*)&Ktile[1][0]};
  const unsigned ldsV[2] = {(unsigned)(size_t)(void*)&VtileT[0][0],
                            (unsigned)(size_t)(void*)&VtileT[1][0]};

  // prologue: start tile 0 DMA
  if (wave == 0) {
    tdm_load_2d(ldsK[0], Kh + base, HDIM, 128, HDIM);
    tdm_load_2d(ldsV[0], VhT + base, 128, HDIM, SEQ);
  }

  for (int kt = 0; kt < SEQ / 128; ++kt) {
    const int cur = kt & 1;
    if (wave == 0) {
      if (kt + 1 < SEQ / 128) {
        // start next tile into the other buffer, then wait for current tile
        tdm_load_2d(ldsK[cur ^ 1], Kh + base + (size_t)(kt + 1) * 128 * HDIM,
                    HDIM, 128, HDIM);
        tdm_load_2d(ldsV[cur ^ 1], VhT + base + (size_t)(kt + 1) * 128,
                    128, HDIM, SEQ);
        __builtin_amdgcn_s_wait_tensorcnt(2);   // in-order: tile kt complete
      } else {
        __builtin_amdgcn_s_wait_tensorcnt(0);
      }
    }
    __syncthreads();
    const _Float16* Kt = &Ktile[cur][0];
    const _Float16* Vt = &VtileT[cur][0];

#pragma unroll
    for (int ch = 0; ch < 4; ++ch) {              // 32 keys per chunk
      const int key0 = ch * 32;
      // B-fragments of K^T (lane = key column, per-lane contiguous head_dim)
      const v8h* kr0 = (const v8h*)(Kt + (key0 + lrow) * HDIM);
      const v8h* kr1 = (const v8h*)(Kt + (key0 + 16 + lrow) * HDIM);
      v16h bk00 = cat8(kr0[hi],     kr0[2 + hi]);
      v16h bk01 = cat8(kr0[4 + hi], kr0[6 + hi]);
      v16h bk10 = cat8(kr1[hi],     kr1[2 + hi]);
      v16h bk11 = cat8(kr1[4 + hi], kr1[6 + hi]);

      v8f s0 = {}, s1 = {};
      s0 = wmma_f16(aq0, bk00, s0);
      s0 = wmma_f16(aq1, bk01, s0);
      s1 = wmma_f16(aq0, bk10, s1);
      s1 = wmma_f16(aq1, bk11, s1);

      // online softmax (row r lives in VGPR r, cols across 16-lane halves)
      float alpha[8];
#pragma unroll
      for (int r = 0; r < 8; ++r) {
        float v = fmaxf(s0[r], s1[r]);
#pragma unroll
        for (int msk = 8; msk >= 1; msk >>= 1)
          v = fmaxf(v, __shfl_xor(v, msk, 32));
        float mn = fmaxf(mrow[r], v);
        alpha[r] = __expf(mrow[r] - mn);
        mrow[r]  = mn;
      }
      _Float16* pl = &Pscr[wave][0];
#pragma unroll
      for (int r = 0; r < 8; ++r) {
        float p0 = __expf(s0[r] - mrow[r]);
        float p1 = __expf(s1[r] - mrow[r]);
        float sum = p0 + p1;
#pragma unroll
        for (int msk = 8; msk >= 1; msk >>= 1)
          sum += __shfl_xor(sum, msk, 32);
        lsum[r] = lsum[r] * alpha[r] + sum;
#pragma unroll
        for (int cb = 0; cb < 4; ++cb) oacc[cb][r] *= alpha[r];
        const int m = r + 8 * hi;
        pl[m * 32 + lrow]      = (_Float16)p0;    // C-layout -> LDS
        pl[m * 32 + 16 + lrow] = (_Float16)p1;
      }
      // P back as A-fragment (16x32): per-lane contiguous b128 reads
      const v8h* pr = (const v8h*)(pl + lrow * 32);
      v16h pa = cat8(pr[hi], pr[2 + hi]);

      // O += P @ V : B-fragments from dim-major VtileT, contiguous per lane
#pragma unroll
      for (int cb = 0; cb < 4; ++cb) {
        const v8h* vr = (const v8h*)(Vt + (cb * 16 + lrow) * 128 + key0);
        v16h bv = cat8(vr[hi], vr[2 + hi]);
        oacc[cb] = wmma_f16(pa, bv, oacc[cb]);
      }
    }
    __syncthreads();    // all waves done with buf[cur] before TDM reuses it
  }

  // normalize by softmax denominator
#pragma unroll
  for (int r = 0; r < 8; ++r) {
    float inv = 1.0f / lsum[r];
#pragma unroll
    for (int cb = 0; cb < 4; ++cb) oacc[cb][r] *= inv;
  }

  // ---- fused output projection: out = O @ Wo^T + bo -----------------------
  // per-wave scratch in Ktile[0]; only this wave writes/reads its region
  _Float16* osc = &Ktile[0][0] + wave * 16 * HDIM;
#pragma unroll
  for (int r = 0; r < 8; ++r) {
    const int m = r + 8 * hi;
#pragma unroll
    for (int cb = 0; cb < 4; ++cb)
      osc[m * HDIM + cb * 16 + lrow] = (_Float16)oacc[cb][r];
  }
  v16h oa0, oa1;
  {
    const v8h* ov = (const v8h*)(osc + lrow * HDIM);
    oa0 = cat8(ov[hi], ov[2 + hi]);
    oa1 = cat8(ov[4 + hi], ov[6 + hi]);
  }
#pragma unroll
  for (int nb = 0; nb < 4; ++nb) {
    const int col = nb * 16 + lrow;
    v16h b0 = *(const v16h*)(WoP + ((0 + nb) * 32 + lane) * 16);
    v16h b1 = *(const v16h*)(WoP + ((4 + nb) * 32 + lane) * 16);
    v8f c = {};
    c = wmma_f16(oa0, b0, c);
    c = wmma_f16(oa1, b1, c);
    const float bb = bob[col];
#pragma unroll
    for (int r = 0; r < 8; ++r) {
      const int m = qrow + r + 8 * hi;
      out[base + (size_t)m * HDIM + col] = c[r] + bb;
    }
  }
}

// ---------------------------------------------------------------------------
extern "C" void kernel_launch(void* const* d_in, const int* in_sizes, int n_in,
                              void* d_out, int out_size, void* d_ws, size_t ws_size,
                              hipStream_t stream) {
  const float* x  = (const float*)d_in[0];
  const float* Wq = (const float*)d_in[1];
  const float* bq = (const float*)d_in[2];
  const float* Wk = (const float*)d_in[3];
  const float* bk = (const float*)d_in[4];
  const float* Wv = (const float*)d_in[5];
  const float* bv = (const float*)d_in[6];
  const float* Wo = (const float*)d_in[7];
  const float* bo = (const float*)d_in[8];
  float* out = (float*)d_out;

  const size_t SZ = (size_t)BH * SEQ * HDIM;   // elements per Q/K/V buffer
  _Float16* Qh  = (_Float16*)d_ws;
  _Float16* Kh  = Qh + SZ;
  _Float16* VhT = Kh + SZ;                     // transposed: [bh][d][s]

  dim3 block(256);
  dim3 grid(BH * (SEQ / 128));
  qkv_project_kernel<<<grid, block, 0, stream>>>(x, Wq, bq, Wk, bk, Wv, bv,
                                                 Qh, Kh, VhT);
  attn_out_kernel<<<grid, block, 0, stream>>>(Qh, Kh, VhT, Wo, bo, out);
}
